// Tower_63127429316779
// MI455X (gfx1250) — matrix operand order repaired and checked by reading, not compiled, not run on hardware
//
#include <hip/hip_runtime.h>
#include <math.h>

// ---------------------------------------------------------------------------
// Tower RNN for MI455X (gfx1250).  B=256, T=128, D=512, S=256.
//   K0: convert W_inp/W_mid/W_hid f32 -> bf16 in workspace.
//   K1: (parallel over 32768 (b,t) rows) L = relu(relu(E Wi^T + bi) Wm^T + bm)
//       via v_wmma_f32_16x16x32_bf16; L f32 in workspace (64MB, L2-resident).
//   K2: recurrence h = tanh(L[t] + h Wh^T + bh), 64 WGs = 16 batch-slabs x 4
//       N-slices. Each WG holds its 128x512 bf16 W_hid slice in LDS (128KB);
//       h exchanged through a double-buffered global bf16 H with a per-slab
//       arrival counter (only the 4 WGs of a slab synchronize).
//   K3: row-wise L2 normalize of final f32 h.
// Workspace (bytes): [0,256K) Wi | [256K,512K) Wm | [512K,1M) Wh |
//   [1M,1M+64M) L | +512K Hg[2] bf16 | +512K Hf f32 | +64B counters.
// ---------------------------------------------------------------------------

#define BATCH 256
#define TSEQ  128
#define DIM   512
#define SHR   256

typedef __attribute__((ext_vector_type(16))) __bf16 bf16x16;
typedef __attribute__((ext_vector_type(8)))  __bf16 bf16x8;
typedef __attribute__((ext_vector_type(8)))  float  f32x8;

__device__ __forceinline__ f32x8 wmma_bf16(bf16x16 a, bf16x16 b, f32x8 c) {
  return __builtin_amdgcn_wmma_f32_16x16x32_bf16(false, a, false, b, (short)0, c,
                                                 false, false);
}

// A-matrix (16x32 bf16, MxK), row-major source, leading dim ld (ISA 7.12.2):
// lanes 0-15 row M=lane hold K=[k0..k0+7],[k0+16..k0+23]; lanes 16-31 +8.
__device__ __forceinline__ bf16x16 load_a16(const __bf16* base, int ld, int k0, int lane) {
  int m  = lane & 15;
  int kk = k0 + ((lane & 16) ? 8 : 0);
  const __bf16* p = base + m * ld + kk;
  bf16x8 lo = *(const bf16x8*)p;
  bf16x8 hi = *(const bf16x8*)(p + 16);
  return __builtin_shufflevector(lo, hi, 0,1,2,3,4,5,6,7,8,9,10,11,12,13,14,15);
}

// B-matrix (32x16 bf16, KxN) with B(k,n)=W[n,k], W row-major over k:
// lane n holds K=[k0..k0+7],[k0+16..k0+23] (lanes 16-31: +8 in K), matching A.
__device__ __forceinline__ bf16x16 load_b16(const __bf16* wrow_base, int ld, int k0, int lane) {
  int n  = lane & 15;
  int kk = k0 + ((lane & 16) ? 8 : 0);
  const __bf16* p = wrow_base + n * ld + kk;
  bf16x8 lo = *(const bf16x8*)p;
  bf16x8 hi = *(const bf16x8*)(p + 16);
  return __builtin_shufflevector(lo, hi, 0,1,2,3,4,5,6,7,8,9,10,11,12,13,14,15);
}

__device__ __forceinline__ unsigned short f2bf_bits(float f) {
  union { __bf16 h; unsigned short u; } cv;
  cv.h = (__bf16)f;
  return cv.u;
}

// ---------------- K0: weight conversion f32 -> bf16 ------------------------
__global__ void convert_weights(const float* __restrict__ Wi,
                                const float* __restrict__ Wm,
                                const float* __restrict__ Wh,
                                unsigned short* __restrict__ Wi_b,
                                unsigned short* __restrict__ Wm_b,
                                unsigned short* __restrict__ Wh_b) {
  int i = blockIdx.x * 256 + threadIdx.x;
  if (i < SHR * DIM) Wi_b[i] = f2bf_bits(Wi[i]);
  if (i < DIM * SHR) Wm_b[i] = f2bf_bits(Wm[i]);
  if (i < DIM * DIM) Wh_b[i] = f2bf_bits(Wh[i]);
}

// ---------------- init: zero H ping buffer + slab counters -----------------
__global__ void init_state(unsigned short* __restrict__ Hg0,
                           unsigned int* __restrict__ cnt) {
  int i = blockIdx.x * 256 + threadIdx.x;
  if (i < BATCH * DIM) Hg0[i] = 0;     // bf16 +0.0
  if (i < 16) cnt[i] = 0;
}

// ---------------- K1: L = relu(relu(E Wi^T + bi) Wm^T + bm) ----------------
// grid.x = TSEQ * (BATCH/16) = 2048, 256 threads (8 waves).
__global__ __launch_bounds__(256)
void precompute_L(const int* __restrict__ x,
                  const float* __restrict__ table,
                  const unsigned short* __restrict__ Wi_bits,
                  const float* __restrict__ bi,
                  const unsigned short* __restrict__ Wm_bits,
                  const float* __restrict__ bm,
                  float* __restrict__ L) {
  __shared__ __bf16 emb_s[16 * DIM];
  __shared__ __bf16 lin_s[16 * SHR];
  __shared__ int    tok_s[16];

  const __bf16* Wi_b = (const __bf16*)Wi_bits;
  const __bf16* Wm_b = (const __bf16*)Wm_bits;

  const int t   = blockIdx.x >> 4;
  const int b0  = (blockIdx.x & 15) * 16;
  const int tid = threadIdx.x;
  const int wave = tid >> 5, lane = tid & 31;

  if (tid < 16) tok_s[tid] = x[(b0 + tid) * TSEQ + t];
  __syncthreads();

  for (int i = tid; i < 16 * DIM; i += 256) {
    int r = i >> 9, c = i & (DIM - 1);
    int tok = tok_s[r];
    float v = (tok == 0) ? 0.0f : table[(long)tok * DIM + c];
    emb_s[i] = (__bf16)v;
  }
  __syncthreads();

  for (int j = 0; j < 2; ++j) {        // GEMM1: 2 N-tiles per wave
    const int nt = 2 * wave + j;
    f32x8 acc = {};
#pragma unroll
    for (int k0 = 0; k0 < DIM; k0 += 32) {
      bf16x16 a = load_a16(emb_s, DIM, k0, lane);
      bf16x16 b = load_b16(Wi_b + nt * 16 * DIM, DIM, k0, lane);
      acc = wmma_bf16(a, b, acc);
    }
    const int col = nt * 16 + (lane & 15);
    const float bias = bi[col];
    const int mbase = (lane & 16) ? 8 : 0;
#pragma unroll
    for (int r = 0; r < 8; ++r) {
      float v = acc[r] + bias;
      v = v > 0.0f ? v : 0.0f;
      lin_s[(mbase + r) * SHR + col] = (__bf16)v;
    }
  }
  __syncthreads();

  float* Lrow = L + ((long)t * BATCH + b0) * DIM;
  for (int j = 0; j < 4; ++j) {        // GEMM2: 4 N-tiles per wave
    const int nt = 4 * wave + j;
    f32x8 acc = {};
#pragma unroll
    for (int k0 = 0; k0 < SHR; k0 += 32) {
      bf16x16 a = load_a16(lin_s, SHR, k0, lane);
      bf16x16 b = load_b16(Wm_b + nt * 16 * SHR, SHR, k0, lane);
      acc = wmma_bf16(a, b, acc);
    }
    const int col = nt * 16 + (lane & 15);
    const float bias = bm[col];
    const int mbase = (lane & 16) ? 8 : 0;
#pragma unroll
    for (int r = 0; r < 8; ++r) {
      float v = acc[r] + bias;
      v = v > 0.0f ? v : 0.0f;
      Lrow[(mbase + r) * DIM + col] = v;
    }
  }
}

// ---------------- K2: recurrence, 64 WGs (16 slabs x 4 N-slices) -----------
// Dynamic LDS: [0,128K) W_hid slice bf16 | [128K,128K+16K) h slab bf16.
__global__ __launch_bounds__(256)
void recurrence(const float* __restrict__ L,
                const unsigned short* __restrict__ Wh_bits,
                const float* __restrict__ bh,
                unsigned short* __restrict__ Hg0,
                unsigned short* __restrict__ Hg1,
                float* __restrict__ Hf,
                unsigned int* __restrict__ cnt) {
  extern __shared__ char smem[];
  __bf16* Wh_lds = (__bf16*)smem;                 // 128 x 512 bf16 = 128 KB
  __bf16* h_s    = (__bf16*)(smem + 128 * DIM * 2); // 16 x 512 bf16 = 16 KB

  const int slab = blockIdx.x >> 2;   // 0..15 : batch slab
  const int ns   = blockIdx.x & 3;    // 0..3  : N slice
  const int b0   = slab * 16;
  const int n0   = ns * 128;
  const int tid  = threadIdx.x;
  const int wave = tid >> 5, lane = tid & 31;
  const int mbase = (lane & 16) ? 8 : 0;

  // Preload this WG's contiguous W_hid slice into LDS (131072 B).
  {
    const uint4* src = (const uint4*)(Wh_bits + (long)n0 * DIM);
    uint4* dst = (uint4*)Wh_lds;
    for (int i = tid; i < (128 * DIM * 2) / 16; i += 256) dst[i] = src[i];
  }
  __syncthreads();

  const __bf16* Bbase = Wh_lds + (wave * 16) * DIM;  // wave owns one 16-col tile
  const int col = n0 + wave * 16 + (lane & 15);
  const float bias = bh[col];

  for (int t = 0; t < TSEQ; ++t) {
    if (t > 0) {                       // wait for all 4 slab WGs' step t-1 writes
      if (tid == 0) {
        const unsigned target = (unsigned)(4 * t);
        while (__hip_atomic_load(&cnt[slab], __ATOMIC_ACQUIRE,
                                 __HIP_MEMORY_SCOPE_AGENT) < target)
          __builtin_amdgcn_s_sleep(2);
      }
      __syncthreads();
    }

    // Copy full-width h slab (16 x 512 bf16) from the read buffer into LDS.
    {
      const unsigned short* Hin = (t & 1) ? Hg1 : Hg0;
      const uint4* src = (const uint4*)(Hin + (long)b0 * DIM);
      uint4* dst = (uint4*)h_s;
      for (int i = tid; i < (16 * DIM * 2) / 16; i += 256) dst[i] = src[i];
    }
    __syncthreads();

    const float* Lrow = L + ((long)t * BATCH + b0) * DIM;
    if (t + 1 < TSEQ)                  // warm next step's L (global_prefetch_b8)
      __builtin_prefetch(L + ((long)(t + 1) * BATCH + b0) * DIM + n0 + tid, 0, 0);

    f32x8 acc = {};
#pragma unroll
    for (int k0 = 0; k0 < DIM; k0 += 32) {
      bf16x16 a = load_a16(h_s, DIM, k0, lane);
      bf16x16 b = load_b16(Bbase, DIM, k0, lane);
      acc = wmma_bf16(a, b, acc);
    }

    unsigned short* Hout = (t & 1) ? Hg0 : Hg1;
#pragma unroll
    for (int r = 0; r < 8; ++r) {
      float v = tanhf(acc[r] + Lrow[(mbase + r) * DIM + col] + bias);
      Hout[(long)(b0 + mbase + r) * DIM + col] = f2bf_bits(v);
      if (t == TSEQ - 1) Hf[(long)(b0 + mbase + r) * DIM + col] = v;
    }

    __threadfence();                   // publish Hout slice at agent scope
    __syncthreads();
    if (tid == 0)
      __hip_atomic_fetch_add(&cnt[slab], 1u, __ATOMIC_RELEASE,
                             __HIP_MEMORY_SCOPE_AGENT);
  }
}

// ---------------- K3: row-wise L2 normalize --------------------------------
__global__ __launch_bounds__(256)
void normalize_rows(const float* __restrict__ Hf, float* __restrict__ out) {
  __shared__ float hf_s[16 * DIM];
  __shared__ float inv_s[16];
  const int b0 = blockIdx.x * 16, tid = threadIdx.x;
  for (int i = tid; i < 16 * DIM; i += 256) hf_s[i] = Hf[(long)b0 * DIM + i];
  __syncthreads();
  if (tid < 16) {
    float s = 0.0f;
    for (int c = 0; c < DIM; ++c) { float v = hf_s[tid * DIM + c]; s += v * v; }
    inv_s[tid] = 1.0f / fmaxf(sqrtf(s), 1e-12f);
  }
  __syncthreads();
  for (int i = tid; i < 16 * DIM; i += 256) {
    int r = i >> 9;
    out[(long)b0 * DIM + i] = hf_s[i] * inv_s[r];
  }
}

// ---------------------------------------------------------------------------
extern "C" void kernel_launch(void* const* d_in, const int* in_sizes, int n_in,
                              void* d_out, int out_size, void* d_ws, size_t ws_size,
                              hipStream_t stream) {
  const int*   x     = (const int*)  d_in[0];
  const float* table = (const float*)d_in[1];
  const float* Wi    = (const float*)d_in[2];
  const float* bi    = (const float*)d_in[3];
  const float* Wm    = (const float*)d_in[4];
  const float* bm    = (const float*)d_in[5];
  const float* Wh    = (const float*)d_in[6];
  const float* bh    = (const float*)d_in[7];
  float* out = (float*)d_out;

  char* ws = (char*)d_ws;
  size_t off = 0;
  unsigned short* Wi_b = (unsigned short*)(ws + off); off += (size_t)SHR * DIM * 2;
  unsigned short* Wm_b = (unsigned short*)(ws + off); off += (size_t)DIM * SHR * 2;
  unsigned short* Wh_b = (unsigned short*)(ws + off); off += (size_t)DIM * DIM * 2;
  float*          Lbuf = (float*)(ws + off);          off += (size_t)TSEQ * BATCH * DIM * 4;
  unsigned short* Hg0  = (unsigned short*)(ws + off); off += (size_t)BATCH * DIM * 2;
  unsigned short* Hg1  = (unsigned short*)(ws + off); off += (size_t)BATCH * DIM * 2;
  float*          Hf   = (float*)(ws + off);          off += (size_t)BATCH * DIM * 4;
  unsigned int*   cnt  = (unsigned int*)(ws + off);

  convert_weights<<<(DIM * DIM + 255) / 256, 256, 0, stream>>>(Wi, Wm, Wh,
                                                               Wi_b, Wm_b, Wh_b);
  init_state<<<(BATCH * DIM + 255) / 256, 256, 0, stream>>>(Hg0, cnt);
  precompute_L<<<TSEQ * (BATCH / 16), 256, 0, stream>>>(x, table, Wi_b, bi,
                                                        Wm_b, bm, Lbuf);
  const size_t rec_lds = (size_t)(128 * DIM * 2 + 16 * DIM * 2);  // 147456 B
  recurrence<<<(BATCH / 16) * 4, 256, rec_lds, stream>>>(Lbuf, Wh_b, bh,
                                                         Hg0, Hg1, Hf, cnt);
  normalize_rows<<<BATCH / 16, 256, 0, stream>>>(Hf, out);
}
